// FeedForwardVTP_57320633533118
// MI455X (gfx1250) — matrix operand order repaired and checked by reading, not compiled, not run on hardware
//
#include <hip/hip_runtime.h>

// FeedForwardVTP fused pipeline for MI455X (gfx1250, wave32, WMMA).
// prune1 -> GEMM1(bf16 WMMA, fused bias+relu+scores2) -> prune2 -> GEMM2(bf16 WMMA, fused bias)

#define NTHREADS 256
constexpr int Bsz = 256;   // batch
constexpr int Np  = 128;   // patches (N)
constexpr int Dch = 1024;  // D
constexpr int Hch = 4096;  // H
constexpr int K1  = 819;   // kept channels before net1
constexpr int K2  = 3276;  // kept channels before net2

typedef __attribute__((ext_vector_type(16))) __bf16 v16bf;
typedef __attribute__((ext_vector_type(8)))  __bf16 v8bf;
typedef __attribute__((ext_vector_type(2)))  __bf16 v2bf;
typedef __attribute__((ext_vector_type(8)))  float  v8f;

// pack two f32 -> packed bf16x2 (round-to-nearest-even), one u32
__device__ inline unsigned bfpack2(float lo, float hi) {
#if __has_builtin(__builtin_amdgcn_cvt_pk_bf16_f32)
  union { v2bf v; unsigned u; } r;
  r.v = __builtin_amdgcn_cvt_pk_bf16_f32(lo, hi);   // single v_cvt_pk_bf16_f32
  return r.u;
#else
  union { float f; unsigned u; } a, b;
  a.f = lo; b.f = hi;
  unsigned ra = (a.u + 0x7FFFu + ((a.u >> 16) & 1u)) >> 16;
  unsigned rb = (b.u + 0x7FFFu + ((b.u >> 16) & 1u)) & 0xFFFF0000u;
  return ra | rb;
#endif
}

// ---------------- bitonic sorts in LDS (NTHREADS threads, S power of two) ----

template<int S>
__device__ void bitonic_score_desc(float* sc, int* id, int tid) {
  // sort by (score desc, idx asc)  == jax.lax.top_k selection order
  for (int k = 2; k <= S; k <<= 1) {
    for (int j = k >> 1; j > 0; j >>= 1) {
      __syncthreads();
      for (int i = tid; i < S; i += NTHREADS) {
        int p = i ^ j;
        if (p > i) {
          float s0 = sc[i], s1 = sc[p];
          int   i0 = id[i], i1 = id[p];
          bool first = (s0 > s1) || (s0 == s1 && i0 < i1); // element i should precede p
          bool up = ((i & k) == 0);
          if (up ? !first : first) { sc[i] = s1; sc[p] = s0; id[i] = i1; id[p] = i0; }
        }
      }
    }
  }
  __syncthreads();
}

template<int S>
__device__ void bitonic_int_asc(int* a, int tid) {
  for (int k = 2; k <= S; k <<= 1) {
    for (int j = k >> 1; j > 0; j >>= 1) {
      __syncthreads();
      for (int i = tid; i < S; i += NTHREADS) {
        int p = i ^ j;
        if (p > i) {
          int A = a[i], C = a[p];
          bool up = ((i & k) == 0);
          if (up ? (A > C) : (A < C)) { a[i] = C; a[p] = A; }
        }
      }
    }
  }
  __syncthreads();
}

// ---------------- prune 1: scores over x + top-K1 ---------------------------

__global__ __launch_bounds__(NTHREADS)
void prune1_kernel(const float* __restrict__ x, const float* __restrict__ ws1,
                   int* __restrict__ idx1) {
  __shared__ float sc[Dch];
  __shared__ int   id[Dch];
  __shared__ float w[Np];
  const int b = blockIdx.x, tid = threadIdx.x;
  if (tid < Np) w[tid] = ws1[tid];
  __syncthreads();
  const float* xb = x + (size_t)b * Np * Dch;
  float a0 = 0.f, a1 = 0.f, a2 = 0.f, a3 = 0.f;
  for (int n = 0; n < Np; ++n) {
    const float wn = w[n];
    const float* r = xb + (size_t)n * Dch;
    a0 = fmaf(r[tid      ], wn, a0);
    a1 = fmaf(r[tid + 256], wn, a1);
    a2 = fmaf(r[tid + 512], wn, a2);
    a3 = fmaf(r[tid + 768], wn, a3);
  }
  sc[tid] = a0; sc[tid + 256] = a1; sc[tid + 512] = a2; sc[tid + 768] = a3;
  id[tid] = tid; id[tid + 256] = tid + 256; id[tid + 512] = tid + 512; id[tid + 768] = tid + 768;
  bitonic_score_desc<Dch>(sc, id, tid);
  for (int i = tid; i < Dch; i += NTHREADS) if (i >= K1) id[i] = 0x7FFFFFFF;
  bitonic_int_asc<Dch>(id, tid);
  for (int i = tid; i < K1; i += NTHREADS) idx1[(size_t)b * K1 + i] = id[i];
}

// ---------------- prune 2: top-K2 of precomputed scores2 --------------------

__global__ __launch_bounds__(NTHREADS)
void prune2_kernel(const float* __restrict__ scores2, int* __restrict__ idx2) {
  __shared__ float sc[Hch];
  __shared__ int   id[Hch];
  const int b = blockIdx.x, tid = threadIdx.x;
  const float* s = scores2 + (size_t)b * Hch;
  for (int i = tid; i < Hch; i += NTHREADS) { sc[i] = s[i]; id[i] = i; }
  bitonic_score_desc<Hch>(sc, id, tid);
  for (int i = tid; i < Hch; i += NTHREADS) if (i >= K2) id[i] = 0x7FFFFFFF;
  bitonic_int_asc<Hch>(id, tid);
  for (int i = tid; i < K2; i += NTHREADS) idx2[(size_t)b * K2 + i] = id[i];
}

// ---------------- gathered GEMM with bf16 WMMA ------------------------------
// Y[b,m,n] = act( sum_k X[b,m,idx[k]] * W[n,k] + bias[n] ), block = 128m x 128n.
// Double-buffered LDS pipeline: global loads for chunk kc+1 issue before the
// WMMAs of chunk kc, so VMEM latency hides behind matrix work.
// Optional fused scores: scores_out[b,n] = sum_m Y[b,m,n] * patch_w[m].

template<int KTOT, int LDX, int HOUT, bool RELU, bool FUSE_SCORES>
__global__ __launch_bounds__(NTHREADS)
void gemm_gather_kernel(const float* __restrict__ X, const int* __restrict__ idx,
                        const float* __restrict__ W, const float* __restrict__ bias,
                        float* __restrict__ Y, float* __restrict__ scores_out,
                        const float* __restrict__ patch_w) {
  constexpr int KCFULL = KTOT / 32;        // guard-free K chunks
  constexpr int KREM   = KTOT % 32;        // tail elements (branchless masked chunk)
  constexpr int KC     = KCFULL + (KREM ? 1 : 0);
  constexpr int LDT    = 40;               // padded LDS row stride (bf16 elems)

  __shared__ __attribute__((aligned(16))) __bf16 At[2][Np  * LDT]; // A: [m][k] row-major
  __shared__ __attribute__((aligned(16))) __bf16 Bt[2][128 * LDT]; // B: [n][k] row-major (W rows)
  __shared__ int   idxs[KTOT];
  __shared__ float scTile[128];
  __shared__ float pw[Np];

  const int tid = threadIdx.x;
  const int b   = blockIdx.y;
  const int hc0 = blockIdx.x * 128;
  const size_t xbase = (size_t)b * Np * LDX;

  for (int i = tid; i < KTOT; i += NTHREADS) idxs[i] = idx[(size_t)b * KTOT + i];
  if constexpr (FUSE_SCORES) {
    if (tid < 128) scTile[tid] = 0.f;
    if (tid < Np)  pw[tid] = patch_w[tid];
  }
  __syncthreads();   // idxs visible before first load_regs

  const int lane  = tid & 31;
  const int wv    = tid >> 5;           // wave id 0..7 -> owns rows [16wv, 16wv+15]
  const int aRow  = wv * 16 + (lane & 15);
  const int kh    = (lane >> 4) * 8;    // A: lanes 0-15 take K {0-7,16-23}; 16-31 take {8-15,24-31}
  const int bRowL = lane & 15;          // B: column N = lane&15
  const int kb    = (lane >> 4) * 8;    // B: same K split as A ({0-7,16-23} / {8-15,24-31})

  v8f acc[8] = {};
  float ra[16], rw[16];                 // in-flight chunk (8 pairs of A and W values)

  // issue global loads for one 32-wide K chunk into registers
  auto load_regs = [&](int kc, bool tail) {
#pragma unroll
    for (int g = 0; g < 8; ++g) {
      const int pi = tid + g * NTHREADS;     // pair index 0..2047
      const int mm = pi >> 4;                // row 0..127
      const int k2 = (pi & 15) * 2;          // even k within chunk
      const int kg = kc * 32 + k2;
      const size_t wrow = (size_t)(hc0 + mm) * KTOT;
      if (!tail) {                            // hot path: no guards
        const int c0 = idxs[kg], c1 = idxs[kg + 1];
        ra[2 * g]     = X[xbase + (size_t)mm * LDX + c0];
        ra[2 * g + 1] = X[xbase + (size_t)mm * LDX + c1];
        rw[2 * g]     = W[wrow + kg];
        rw[2 * g + 1] = W[wrow + kg + 1];
      } else {                                // branchless masked tail, clamped addresses
        const bool i0 = (kg < KTOT), i1 = (kg + 1 < KTOT);
        const int kg0 = i0 ? kg : KTOT - 1;
        const int kg1 = i1 ? kg + 1 : KTOT - 1;
        const int c0 = idxs[kg0], c1 = idxs[kg1];
        const float a0 = X[xbase + (size_t)mm * LDX + c0];
        const float a1 = X[xbase + (size_t)mm * LDX + c1];
        const float w0 = W[wrow + kg0];
        const float w1 = W[wrow + kg1];
        ra[2 * g]     = i0 ? a0 : 0.f;
        ra[2 * g + 1] = i1 ? a1 : 0.f;
        rw[2 * g]     = i0 ? w0 : 0.f;
        rw[2 * g + 1] = i1 ? w1 : 0.f;
      }
    }
  };

  // pack registers as bf16x2 and store to LDS buffer
  auto store_regs = [&](int buf) {
#pragma unroll
    for (int g = 0; g < 8; ++g) {
      const int pi = tid + g * NTHREADS;
      const int mm = pi >> 4;
      const int k2 = (pi & 15) * 2;
      *(unsigned*)&At[buf][mm * LDT + k2] = bfpack2(ra[2 * g], ra[2 * g + 1]);
      *(unsigned*)&Bt[buf][mm * LDT + k2] = bfpack2(rw[2 * g], rw[2 * g + 1]);
    }
  };

  // consume a staged chunk: batch all fragment loads, then 8 WMMAs
  auto compute = [&](int buf) {
    union BF { v16bf v; v8bf h[2]; };
    BF af, bfr[8];
    af.h[0] = *(const v8bf*)&At[buf][aRow * LDT + kh];
    af.h[1] = *(const v8bf*)&At[buf][aRow * LDT + kh + 16];
#pragma unroll
    for (int ct = 0; ct < 8; ++ct) {
      const int br = ct * 16 + bRowL;
      bfr[ct].h[0] = *(const v8bf*)&Bt[buf][br * LDT + kb];
      bfr[ct].h[1] = *(const v8bf*)&Bt[buf][br * LDT + kb + 16];
    }
#pragma unroll
    for (int ct = 0; ct < 8; ++ct)
      acc[ct] = __builtin_amdgcn_wmma_f32_16x16x32_bf16(
          false, af.v, false, bfr[ct].v, (short)0, acc[ct], false, false);
  };

  // --- software-pipelined main loop (double-buffered LDS) ---
  load_regs(0, KCFULL == 0 && KREM != 0);
  store_regs(0);
  __syncthreads();
  for (int kc = 0; kc < KC; ++kc) {
    const int cur = kc & 1;
    const bool more = (kc + 1 < KC);
    if (more) {
      load_regs(kc + 1, (kc + 1 == KCFULL) && KREM != 0);  // VMEM in flight during WMMAs
      // gfx1250 global_prefetch_b8 two chunks ahead on the weight stream
      __builtin_prefetch(&W[(size_t)(hc0 + (tid & 127)) * KTOT + (size_t)(kc + 2) * 32], 0, 1);
    }
    compute(cur);
    if (more) store_regs(cur ^ 1);
    __syncthreads();
  }

  // --- epilogue: bias (+relu), store, optional fused patch-weighted scores ---
  const int halo = lane >> 4;
  const int nlo  = lane & 15;
#pragma unroll
  for (int ct = 0; ct < 8; ++ct) {
    const int n = hc0 + ct * 16 + nlo;
    const float bv = bias[n];
    float part = 0.f;
#pragma unroll
    for (int v = 0; v < 8; ++v) {
      const int m = wv * 16 + halo * 8 + v;   // C/D layout: M = v + 8*(lane>>4)
      float val = acc[ct][v] + bv;
      if (RELU) val = fmaxf(val, 0.f);
      Y[((size_t)b * Np + m) * HOUT + n] = val;
      if constexpr (FUSE_SCORES) part = fmaf(val, pw[m], part);
    }
    if constexpr (FUSE_SCORES)
      atomicAdd(&scTile[ct * 16 + nlo], part);  // ds_add_f32, block-local, 1 per (ct,lane)
  }
  if constexpr (FUSE_SCORES) {
    __syncthreads();
    if (tid < 128) scores_out[(size_t)b * HOUT + hc0 + tid] = scTile[tid];
  }
}

// ---------------- host-side launch ------------------------------------------

extern "C" void kernel_launch(void* const* d_in, const int* in_sizes, int n_in,
                              void* d_out, int out_size, void* d_ws, size_t ws_size,
                              hipStream_t stream) {
  (void)in_sizes; (void)n_in; (void)out_size; (void)ws_size;
  const float* x   = (const float*)d_in[0];
  const float* ws1 = (const float*)d_in[1];
  // d_in[2] = bs1: uniform shift of scores1 -> cannot change top-k; exactly ignorable
  const float* W1  = (const float*)d_in[3];
  const float* b1  = (const float*)d_in[4];
  const float* ws2 = (const float*)d_in[5];
  // d_in[6] = bs2: same, ignorable
  const float* W2  = (const float*)d_in[7];
  const float* b2  = (const float*)d_in[8];
  float* out = (float*)d_out;

  char* p = (char*)d_ws;
  auto a256 = [](size_t v) { return (v + 255) & ~(size_t)255; };
  size_t off = 0;
  int*   idx1    = (int*)(p + off);   off += a256((size_t)Bsz * K1 * sizeof(int));
  int*   idx2    = (int*)(p + off);   off += a256((size_t)Bsz * K2 * sizeof(int));
  float* scores2 = (float*)(p + off); off += a256((size_t)Bsz * Hch * sizeof(float));
  float* h       = (float*)(p + off); // Bsz*Np*Hch f32 (~537 MB)

  // 1) channel scores on x + top-K1 (exact f32)
  prune1_kernel<<<Bsz, NTHREADS, 0, stream>>>(x, ws1, idx1);

  // 2) GEMM1: h = relu(x[:, :, idx1] @ W1^T + b1); fused scores2 accumulation
  gemm_gather_kernel<K1, Dch, Hch, true, true>
      <<<dim3(Hch / 128, Bsz), NTHREADS, 0, stream>>>(x, idx1, W1, b1, h, scores2, ws2);

  // 3) top-K2 on scores2
  prune2_kernel<<<Bsz, NTHREADS, 0, stream>>>(scores2, idx2);

  // 4) GEMM2: out = h[:, :, idx2] @ W2^T + b2
  gemm_gather_kernel<K2, Hch, Dch, false, false>
      <<<dim3(Dch / 128, Bsz), NTHREADS, 0, stream>>>(h, idx2, W2, b2, out, nullptr, nullptr);
}
